// LEAD_22308060135935
// MI455X (gfx1250) — matrix operand order repaired and checked by reading, not compiled, not run on hardware
//
#include <hip/hip_runtime.h>
#include <math.h>

typedef __bf16 bf16;
typedef bf16  v16bf __attribute__((ext_vector_type(16)));
typedef float v8f   __attribute__((ext_vector_type(8)));

#define NLAY 12
#define DMODEL 512
#define NHEAD 8
#define DHEAD 64
#define NEXP 4
#define FDIM 1024
#define NB 2
#define LSEQ 2048
#define NTOK (NB*LSEQ)          /* 4096 */
#define LN_EPS 1e-5f
#define LSH_EPS 1e-4f

// MoE tiling: 32 tokens per block (2 x 16-row WMMA sub-tiles)
#define MTILE 32
#define TILES_PER_E (NTOK / MTILE)      /* 128 */
// LDS strides chosen so (stride_words/4) % 16 == 3 -> conflict-free ds_load_b128
#define LDA 536                          /* bf16 elems, 1072 B */
#define LDH 1048                         /* bf16 elems, 2096 B */
#define SM_A_BYTES   (MTILE * LDA * 2)                  /* 34304  */
#define SM_H_OFF     SM_A_BYTES
#define SM_H_BYTES   (MTILE * LDH * 2)                  /* 67072  */
#define SM_TOK_OFF   (SM_H_OFF + SM_H_BYTES)            /* 101376 */
#define SM_WSL_OFF   (SM_TOK_OFF + 128)
#define SM_TOTAL     (SM_WSL_OFF + 128)                 /* 101632 */

// ---------------------------------------------------------------------------
// x1[.,d] = x[.,d/2]; x2[.,d] = x[., D/2 + d/2]   (repeat(2,axis=2) + split)
// ---------------------------------------------------------------------------
__global__ void k_init(const float* __restrict__ x,
                       float* __restrict__ x1, float* __restrict__ x2) {
  int i = blockIdx.x * blockDim.x + threadIdx.x;
  if (i >= NTOK * DMODEL) return;
  int d   = i & (DMODEL - 1);
  int tok = i >> 9;
  const float* row = x + (size_t)tok * DMODEL;
  x1[i] = row[d >> 1];
  x2[i] = row[256 + (d >> 1)];
}

// ---------------------------------------------------------------------------
// LayerNorm(x2) -> t ; hash proj (H,2) ; angles (B,H,L) for sorting
// ---------------------------------------------------------------------------
__global__ void k_ln1_hash(const float* __restrict__ xin,
                           const float* __restrict__ g, const float* __restrict__ be,
                           const float* __restrict__ hw, const float* __restrict__ hb,
                           float* __restrict__ t, float* __restrict__ angles, int layer) {
  __shared__ float row[DMODEL];
  __shared__ float red[256];
  __shared__ float proj[16];
  __shared__ float s_mu, s_rstd;
  int tok = blockIdx.x, tid = threadIdx.x;
  size_t base = (size_t)tok * DMODEL;
  float a0 = xin[base + tid], a1 = xin[base + 256 + tid];
  red[tid] = a0 + a1; __syncthreads();
  for (int s = 128; s > 0; s >>= 1) { if (tid < s) red[tid] += red[tid + s]; __syncthreads(); }
  if (tid == 0) s_mu = red[0] * (1.f / DMODEL);
  __syncthreads();
  float mu = s_mu;
  float d0 = a0 - mu, d1 = a1 - mu;
  red[tid] = d0 * d0 + d1 * d1; __syncthreads();
  for (int s = 128; s > 0; s >>= 1) { if (tid < s) red[tid] += red[tid + s]; __syncthreads(); }
  if (tid == 0) s_rstd = rsqrtf(red[0] * (1.f / DMODEL) + LN_EPS);
  __syncthreads();
  float rs = s_rstd;
  const float* gp = g + (size_t)layer * DMODEL;
  const float* bp = be + (size_t)layer * DMODEL;
  float t0 = d0 * rs * gp[tid]       + bp[tid];
  float t1 = d1 * rs * gp[tid + 256] + bp[tid + 256];
  t[base + tid] = t0; t[base + 256 + tid] = t1;
  row[tid] = t0; row[tid + 256] = t1;
  __syncthreads();
  if (tid < 16) {                       // proj[h*2+k]
    int h = tid >> 1, k = tid & 1;
    const float* w = hw + (((size_t)layer * NHEAD + h) * DHEAD) * 2 + k;
    float s = hb[((size_t)layer * NHEAD + h) * 2 + k];
    const float* rr = row + h * DHEAD;
    for (int d = 0; d < DHEAD; ++d) s += rr[d] * w[2 * d];
    proj[tid] = s;
  }
  __syncthreads();
  if (tid < NHEAD) {                    // h_x = proj[0:8], h_y = proj[8:16]
    float ang = atanf(proj[tid] / (proj[8 + tid] + LSH_EPS));
    int b = tok / LSEQ, l = tok % LSEQ;
    angles[((size_t)(b * NHEAD + tid)) * LSEQ + l] = ang;
  }
}

// ---------------------------------------------------------------------------
// Bitonic argsort of 2048 angles per (b,h).  idx[rank] = original token pos.
// ---------------------------------------------------------------------------
__global__ void k_sort(const float* __restrict__ angles, int* __restrict__ idx) {
  __shared__ float key[LSEQ];
  __shared__ int   val[LSEQ];
  int bh = blockIdx.x, tid = threadIdx.x;
  const float* ap = angles + (size_t)bh * LSEQ;
  for (int i = tid; i < LSEQ; i += 1024) { key[i] = ap[i]; val[i] = i; }
  __syncthreads();
  for (int k = 2; k <= LSEQ; k <<= 1)
    for (int j = k >> 1; j > 0; j >>= 1) {
      for (int i = tid; i < LSEQ; i += 1024) {
        int ixj = i ^ j;
        if (ixj > i) {
          bool asc = ((i & k) == 0);
          float ka = key[i], kb = key[ixj];
          bool sw = asc ? (ka > kb) : (ka < kb);
          if (sw) {
            key[i] = kb; key[ixj] = ka;
            int v = val[i]; val[i] = val[ixj]; val[ixj] = v;
          }
        }
      }
      __syncthreads();
    }
  int* op = idx + (size_t)bh * LSEQ;
  for (int i = tid; i < LSEQ; i += 1024) op[i] = val[i];
}

// ---------------------------------------------------------------------------
// gather(sorted) -> grouped conv (64 in/64 out per head, 3 taps, wrap pad)
// -> scatter-add into x1 (y1 = x1 + out).  64-row chunks per block.
// ---------------------------------------------------------------------------
__global__ void k_conv(const float* __restrict__ t, const int* __restrict__ idx,
                       const float* __restrict__ cw, const float* __restrict__ cb,
                       float* x1buf, int layer) {
  __shared__ float xs[66 * 68];         // padded: rows l-1..l+64
  int blk = blockIdx.x;
  int chunk = blk & 31;                 // L/64 = 32 chunks
  int bh = blk >> 5;
  int b = bh >> 3, h = bh & 7;
  int tid = threadIdx.x;
  const int* ip = idx + (size_t)bh * LSEQ;
  for (int i = tid; i < 66 * 64; i += 256) {
    int r = i >> 6, c = i & 63;
    int l = (chunk * 64 + r - 1 + LSEQ) & (LSEQ - 1);   // wrap
    int pos = ip[l];
    xs[r * 68 + c] = t[((size_t)(b * LSEQ + pos)) * DMODEL + h * DHEAD + c];
  }
  __syncthreads();
  int r = tid >> 2, oq = tid & 3, o0 = oq * 16;
  const float* cwp = cw + ((size_t)layer * DMODEL + h * DHEAD + o0) * (DHEAD * 3);
  float acc[16];
#pragma unroll
  for (int o = 0; o < 16; ++o) acc[o] = 0.f;
  for (int i = 0; i < DHEAD; ++i) {
    float v0 = xs[r * 68 + i], v1 = xs[(r + 1) * 68 + i], v2 = xs[(r + 2) * 68 + i];
#pragma unroll 4
    for (int o = 0; o < 16; ++o) {
      const float* w = cwp + o * (DHEAD * 3) + i * 3;
      acc[o] += v0 * w[0] + v1 * w[1] + v2 * w[2];
    }
  }
  int l = chunk * 64 + r;
  int pos = ip[l];
  size_t base = ((size_t)(b * LSEQ + pos)) * DMODEL + h * DHEAD + o0;
  const float* cbp = cb + (size_t)layer * DMODEL + h * DHEAD + o0;
#pragma unroll
  for (int o = 0; o < 16; ++o) x1buf[base + o] += acc[o] + cbp[o];
}

// ---------------------------------------------------------------------------
// LayerNorm(y1) -> t2 (bf16) ; gate softmax -> top-1 routing lists + wsel
// ---------------------------------------------------------------------------
__global__ void k_ln2_gate(const float* __restrict__ xin,
                           const float* __restrict__ g, const float* __restrict__ be,
                           const float* __restrict__ gw, const float* __restrict__ gb,
                           bf16* __restrict__ t2, float* __restrict__ wselbuf,
                           int* counts, int* lists, int layer) {
  __shared__ float red[256];
  __shared__ float gred[NEXP * 256];
  __shared__ float s_mu, s_rstd;
  int tok = blockIdx.x, tid = threadIdx.x;
  size_t base = (size_t)tok * DMODEL;
  float a0 = xin[base + tid], a1 = xin[base + 256 + tid];
  red[tid] = a0 + a1; __syncthreads();
  for (int s = 128; s > 0; s >>= 1) { if (tid < s) red[tid] += red[tid + s]; __syncthreads(); }
  if (tid == 0) s_mu = red[0] * (1.f / DMODEL);
  __syncthreads();
  float mu = s_mu;
  float d0 = a0 - mu, d1 = a1 - mu;
  red[tid] = d0 * d0 + d1 * d1; __syncthreads();
  for (int s = 128; s > 0; s >>= 1) { if (tid < s) red[tid] += red[tid + s]; __syncthreads(); }
  if (tid == 0) s_rstd = rsqrtf(red[0] * (1.f / DMODEL) + LN_EPS);
  __syncthreads();
  float rs = s_rstd;
  const float* gp = g + (size_t)layer * DMODEL;
  const float* bp = be + (size_t)layer * DMODEL;
  float t0 = d0 * rs * gp[tid]       + bp[tid];
  float t1 = d1 * rs * gp[tid + 256] + bp[tid + 256];
  t2[base + tid]       = (bf16)t0;
  t2[base + 256 + tid] = (bf16)t1;
  const float* gwp = gw + (size_t)layer * DMODEL * NEXP;
#pragma unroll
  for (int e = 0; e < NEXP; ++e)
    gred[e * 256 + tid] = t0 * gwp[tid * NEXP + e] + t1 * gwp[(tid + 256) * NEXP + e];
  __syncthreads();
  for (int s = 128; s > 0; s >>= 1) {
    if (tid < s)
      for (int e = 0; e < NEXP; ++e) gred[e * 256 + tid] += gred[e * 256 + tid + s];
    __syncthreads();
  }
  if (tid == 0) {
    float lg[NEXP];
    for (int e = 0; e < NEXP; ++e) lg[e] = gred[e * 256] + gb[layer * NEXP + e];
    float m = lg[0]; int top = 0;
    for (int e = 1; e < NEXP; ++e) if (lg[e] > m) { m = lg[e]; top = e; }
    float den = 0.f;
    for (int e = 0; e < NEXP; ++e) den += expf(lg[e] - m);
    wselbuf[tok] = 1.f / den;                 // softmax prob of the max
    int slot = atomicAdd(&counts[top], 1);
    lists[top * NTOK + slot] = tok;
  }
}

// ---------------------------------------------------------------------------
// Per-layer weight transpose + f32->bf16:  w1t[e][f][d], w2t[e][d][f]
// ---------------------------------------------------------------------------
__global__ void k_cvt(const float* __restrict__ w1, const float* __restrict__ w2,
                      bf16* __restrict__ w1t, bf16* __restrict__ w2t, int layer) {
  const size_t EDF = (size_t)NEXP * DMODEL * FDIM;
  size_t i = (size_t)blockIdx.x * blockDim.x + threadIdx.x;
  if (i < EDF) {
    size_t e = i / ((size_t)FDIM * DMODEL);
    size_t r = i % ((size_t)FDIM * DMODEL);
    size_t f = r / DMODEL, d = r % DMODEL;
    w1t[i] = (bf16)w1[(((size_t)layer * NEXP + e) * DMODEL + d) * FDIM + f];
  } else if (i < 2 * EDF) {
    size_t j = i - EDF;
    size_t e = j / ((size_t)DMODEL * FDIM);
    size_t r = j % ((size_t)DMODEL * FDIM);
    size_t n = r / FDIM, k = r % FDIM;
    w2t[j] = (bf16)w2[(((size_t)layer * NEXP + e) * FDIM + k) * DMODEL + n];
  }
}

__global__ void k_zero(int* counts) { if (threadIdx.x < NEXP) counts[threadIdx.x] = 0; }

// ---------------------------------------------------------------------------
// MoE expert FFN with WMMA bf16.  Block = 32 tokens of one expert, 8 waves.
// Register tile 2x2 (2 m-subtiles x 2 n-tiles): each A fragment feeds 2 wmma
// (1 ds_load_b128/wmma) and each B fragment feeds 2 wmma (1 global b128/wmma).
// GEMM1: (32x512)x(512x1024) -> relu -> h(LDS) ; GEMM2: (32x1024)x(1024x512)
// x2 += wsel * y.   Dynamic LDS ~99 KB (gfx1250 WGP has 320 KB).
// ---------------------------------------------------------------------------
__global__ void __launch_bounds__(256)
k_moe(const bf16* __restrict__ t2,
      const bf16* __restrict__ w1t, const bf16* __restrict__ w2t,
      const float* __restrict__ b1g, const float* __restrict__ b2g,
      const float* __restrict__ wselbuf,
      const int* __restrict__ counts, const int* __restrict__ lists,
      float* x2buf, int layer) {
  extern __shared__ char smem[];
  bf16*  aA   = (bf16*)smem;
  bf16*  hh   = (bf16*)(smem + SM_H_OFF);
  int*   toks = (int*)(smem + SM_TOK_OFF);
  float* wsl  = (float*)(smem + SM_WSL_OFF);

  int e    = blockIdx.x / TILES_PER_E;
  int tile = blockIdx.x % TILES_PER_E;
  int cnt  = counts[e];
  if (tile * MTILE >= cnt) return;
  int tid = threadIdx.x;
  if (tid < MTILE) {
    int gi = tile * MTILE + tid;
    int tk = (gi < cnt) ? lists[e * NTOK + gi] : -1;
    toks[tid] = tk;
    wsl[tid]  = (tk >= 0) ? wselbuf[tk] : 0.f;
  }
  __syncthreads();
  for (int i = tid; i < MTILE * 64; i += 256) {   // 64 x uint4 per row
    int r = i >> 6, c8 = i & 63;
    uint4* dst = (uint4*)&aA[r * LDA + c8 * 8];
    int tk = toks[r];
    if (tk >= 0) *dst = *(const uint4*)(t2 + (size_t)tk * DMODEL + c8 * 8);
    else         *dst = make_uint4(0u, 0u, 0u, 0u);
  }
  __syncthreads();

  int lane = tid & 31, wave = tid >> 5;
  int half = lane >> 4, ln = lane & 15;
  const bf16* w1e = w1t + (size_t)e * FDIM * DMODEL;
  const bf16* w2e = w2t + (size_t)e * DMODEL * FDIM;
  const float* b1p = b1g + ((size_t)layer * NEXP + e) * FDIM;
  const float* b2p = b2g + ((size_t)layer * NEXP + e) * DMODEL;

  // ---- GEMM1: K=512, N=1024 (32 n-tile pairs, 4 per wave; 2 m-subtiles) ----
  for (int ntp = wave; ntp < 32; ntp += 8) {
    int n0 = (2 * ntp) * 16 + ln;
    int n1 = n0 + 16;
    const bf16* bcol0 = w1e + (size_t)n0 * DMODEL + half * 16;
    const bf16* bcol1 = bcol0 + (size_t)16 * DMODEL;
    const bf16* arow0 = aA + ln * LDA + half * 16;
    const bf16* arow1 = arow0 + 16 * LDA;
    v8f acc00 = {0.f, 0.f, 0.f, 0.f, 0.f, 0.f, 0.f, 0.f};
    v8f acc01 = acc00, acc10 = acc00, acc11 = acc00;
#pragma unroll 2
    for (int kk = 0; kk < DMODEL; kk += 32) {
      union { uint4 u[2]; v16bf v; } a0, a1;
      a0.u[0] = *(const uint4*)(arow0 + kk);            // K = kk+16*half+0..7
      a0.u[1] = *(const uint4*)(arow0 + kk + 8);        // K = kk+16*half+8..15
      a1.u[0] = *(const uint4*)(arow1 + kk);
      a1.u[1] = *(const uint4*)(arow1 + kk + 8);
      v16bf bv0 = *(const v16bf*)(bcol0 + kk);          // K = kk+16*half+0..15
      v16bf bv1 = *(const v16bf*)(bcol1 + kk);
      acc00 = __builtin_amdgcn_wmma_f32_16x16x32_bf16(false, a0.v, false, bv0,
                                                      (short)0, acc00, false, false);
      acc10 = __builtin_amdgcn_wmma_f32_16x16x32_bf16(false, a1.v, false, bv0,
                                                      (short)0, acc10, false, false);
      acc01 = __builtin_amdgcn_wmma_f32_16x16x32_bf16(false, a0.v, false, bv1,
                                                      (short)0, acc01, false, false);
      acc11 = __builtin_amdgcn_wmma_f32_16x16x32_bf16(false, a1.v, false, bv1,
                                                      (short)0, acc11, false, false);
    }
    float bias0 = b1p[n0], bias1 = b1p[n1];
#pragma unroll
    for (int rr = 0; rr < 8; ++rr) {
      int m = rr + 8 * half;
      hh[m * LDH + n0]        = (bf16)fmaxf(acc00[rr] + bias0, 0.f);
      hh[(16 + m) * LDH + n0] = (bf16)fmaxf(acc10[rr] + bias0, 0.f);
      hh[m * LDH + n1]        = (bf16)fmaxf(acc01[rr] + bias1, 0.f);
      hh[(16 + m) * LDH + n1] = (bf16)fmaxf(acc11[rr] + bias1, 0.f);
    }
  }
  __syncthreads();

  // ---- GEMM2: K=1024, N=512 (16 n-tile pairs, 2 per wave; 2 m-subtiles) ----
  for (int ntp = wave; ntp < 16; ntp += 8) {
    int n0 = (2 * ntp) * 16 + ln;
    int n1 = n0 + 16;
    const bf16* bcol0 = w2e + (size_t)n0 * FDIM + half * 16;
    const bf16* bcol1 = bcol0 + (size_t)16 * FDIM;
    const bf16* arow0 = hh + ln * LDH + half * 16;
    const bf16* arow1 = arow0 + 16 * LDH;
    v8f acc00 = {0.f, 0.f, 0.f, 0.f, 0.f, 0.f, 0.f, 0.f};
    v8f acc01 = acc00, acc10 = acc00, acc11 = acc00;
#pragma unroll 2
    for (int kk = 0; kk < FDIM; kk += 32) {
      union { uint4 u[2]; v16bf v; } a0, a1;
      a0.u[0] = *(const uint4*)(arow0 + kk);
      a0.u[1] = *(const uint4*)(arow0 + kk + 8);
      a1.u[0] = *(const uint4*)(arow1 + kk);
      a1.u[1] = *(const uint4*)(arow1 + kk + 8);
      v16bf bv0 = *(const v16bf*)(bcol0 + kk);
      v16bf bv1 = *(const v16bf*)(bcol1 + kk);
      acc00 = __builtin_amdgcn_wmma_f32_16x16x32_bf16(false, a0.v, false, bv0,
                                                      (short)0, acc00, false, false);
      acc10 = __builtin_amdgcn_wmma_f32_16x16x32_bf16(false, a1.v, false, bv0,
                                                      (short)0, acc10, false, false);
      acc01 = __builtin_amdgcn_wmma_f32_16x16x32_bf16(false, a0.v, false, bv1,
                                                      (short)0, acc01, false, false);
      acc11 = __builtin_amdgcn_wmma_f32_16x16x32_bf16(false, a1.v, false, bv1,
                                                      (short)0, acc11, false, false);
    }
    float bias0 = b2p[n0], bias1 = b2p[n1];
#pragma unroll
    for (int rr = 0; rr < 8; ++rr) {
      int m = rr + 8 * half;
      int tk0 = toks[m], tk1 = toks[16 + m];
      if (tk0 >= 0) {
        x2buf[(size_t)tk0 * DMODEL + n0] += wsl[m] * (acc00[rr] + bias0);
        x2buf[(size_t)tk0 * DMODEL + n1] += wsl[m] * (acc01[rr] + bias1);
      }
      if (tk1 >= 0) {
        x2buf[(size_t)tk1 * DMODEL + n0] += wsl[16 + m] * (acc10[rr] + bias0);
        x2buf[(size_t)tk1 * DMODEL + n1] += wsl[16 + m] * (acc11[rr] + bias1);
      }
    }
  }
}

__global__ void k_out(const float* __restrict__ x1, const float* __restrict__ x2,
                      float* __restrict__ out) {
  int i = blockIdx.x * blockDim.x + threadIdx.x;
  if (i < NTOK * DMODEL) out[i] = 0.5f * (x1[i] + x2[i]);
}

// ---------------------------------------------------------------------------
extern "C" void kernel_launch(void* const* d_in, const int* in_sizes, int n_in,
                              void* d_out, int out_size, void* d_ws, size_t ws_size,
                              hipStream_t stream) {
  (void)in_sizes; (void)n_in; (void)out_size; (void)ws_size;
  const float* x      = (const float*)d_in[0];
  const float* ln1_g  = (const float*)d_in[1];
  const float* ln1_b  = (const float*)d_in[2];
  const float* ln2_g  = (const float*)d_in[3];
  const float* ln2_b  = (const float*)d_in[4];
  const float* hash_w = (const float*)d_in[5];
  const float* hash_b = (const float*)d_in[6];
  const float* conv_w = (const float*)d_in[7];
  const float* conv_b = (const float*)d_in[8];
  const float* gate_w = (const float*)d_in[9];
  const float* gate_b = (const float*)d_in[10];
  const float* w1     = (const float*)d_in[11];
  const float* b1     = (const float*)d_in[12];
  const float* w2     = (const float*)d_in[13];
  const float* b2     = (const float*)d_in[14];

  char* ws = (char*)d_ws;
  size_t off = 0;
  auto alloc = [&](size_t bytes) -> void* {
    void* p = ws + off;
    off = (off + bytes + 255) & ~(size_t)255;
    return p;
  };
  float* x1buf  = (float*)alloc((size_t)NTOK * DMODEL * 4);
  float* x2buf  = (float*)alloc((size_t)NTOK * DMODEL * 4);
  float* tbuf   = (float*)alloc((size_t)NTOK * DMODEL * 4);
  bf16*  t2buf  = (bf16*) alloc((size_t)NTOK * DMODEL * 2);
  float* angles = (float*)alloc((size_t)NB * NHEAD * LSEQ * 4);
  int*   idxbuf = (int*)  alloc((size_t)NB * NHEAD * LSEQ * 4);
  float* wselb  = (float*)alloc((size_t)NTOK * 4);
  int*   counts = (int*)  alloc((size_t)NEXP * 4);
  int*   lists  = (int*)  alloc((size_t)NEXP * NTOK * 4);
  bf16*  w1t    = (bf16*) alloc((size_t)NEXP * DMODEL * FDIM * 2);
  bf16*  w2t    = (bf16*) alloc((size_t)NEXP * DMODEL * FDIM * 2);

  // allow ~99 KB dynamic LDS for the MoE kernel (gfx1250 WGP: 320 KB)
  (void)hipFuncSetAttribute((const void*)k_moe,
                            hipFuncAttributeMaxDynamicSharedMemorySize, SM_TOTAL);

  k_init<<<(NTOK * DMODEL) / 256, 256, 0, stream>>>(x, x1buf, x2buf);

  for (int l = 0; l < NLAY; ++l) {
    k_cvt<<<(2 * NEXP * DMODEL * FDIM) / 256, 256, 0, stream>>>(w1, w2, w1t, w2t, l);
    k_ln1_hash<<<NTOK, 256, 0, stream>>>(x2buf, ln1_g, ln1_b, hash_w, hash_b,
                                         tbuf, angles, l);
    k_sort<<<NB * NHEAD, 1024, 0, stream>>>(angles, idxbuf);
    k_conv<<<NB * NHEAD * 32, 256, 0, stream>>>(tbuf, idxbuf, conv_w, conv_b, x1buf, l);
    k_zero<<<1, 32, 0, stream>>>(counts);
    k_ln2_gate<<<NTOK, 256, 0, stream>>>(x1buf, ln2_g, ln2_b, gate_w, gate_b,
                                         t2buf, wselb, counts, lists, l);
    k_moe<<<NEXP * TILES_PER_E, 256, SM_TOTAL, stream>>>(t2buf, w1t, w2t, b1, b2,
                                                         wselb, counts, lists, x2buf, l);
  }

  k_out<<<(NTOK * DMODEL + 255) / 256, 256, 0, stream>>>(x1buf, x2buf, (float*)d_out);
}
